// WeierstrassViTL16_4423816315539
// MI455X (gfx1250) — matrix operand order repaired and checked by reading, not compiled, not run on hardware
//
#include <hip/hip_runtime.h>

typedef __attribute__((ext_vector_type(16))) _Float16 v16h;
typedef __attribute__((ext_vector_type(8)))  _Float16 v8h;
typedef __attribute__((ext_vector_type(8)))  float    v8f;
typedef unsigned int v4u __attribute__((ext_vector_type(4)));
typedef unsigned int v8u __attribute__((ext_vector_type(8)));

#define OMEGA1 2.6220575542921198f
#define PI_F   3.14159265358979323846f

__device__ __forceinline__ float softplus_f(float x) {
    return (x > 20.0f) ? x : log1pf(expf(x));
}
__device__ __forceinline__ float gelu_exact(float x) {
    return 0.5f * x * (1.0f + erff(x * 0.70710678118654752f));
}

// ---- CDNA5 async global->LDS copy (ASYNCcnt) --------------------------------
__device__ __forceinline__ void async_ld_b128(unsigned lds_byte_off, const _Float16* gsrc) {
    asm volatile("global_load_async_to_lds_b128 %0, %1, off"
                 :: "v"(lds_byte_off), "v"(gsrc) : "memory");
}
__device__ __forceinline__ void wait_async0() {
#if defined(__has_builtin)
#if __has_builtin(__builtin_amdgcn_s_wait_asynccnt)
    __builtin_amdgcn_s_wait_asynccnt(0);
    return;
#endif
#endif
    asm volatile("s_wait_asynccnt 0x0" ::: "memory");
}

// ---- CDNA5 Tensor Data Mover (TENSORcnt) ------------------------------------
// Builds a D# per CDNA5 ISA and issues TENSOR_LOAD_TO_LDS
// (VIMAGE, 4 SGPR descriptor groups). 2D tile: tile_d0 elems/row, tile_d1 rows,
// row stride stride_d0 (data_size units, 2B). Optional LDS pad engine.
__device__ __forceinline__ void tdm_load_to_lds(
        unsigned lds_addr, const void* gaddr,
        unsigned tile_d0, unsigned tile_d1,
        unsigned tensor_d0, unsigned tensor_d1,
        unsigned stride_d0,
        unsigned pad_en, unsigned pad_interval, unsigned pad_amount)
{
    unsigned long long ga = (unsigned long long)gaddr;
    v4u g0;
    g0.x = 1u;                                                     // count=1 (valid D#)
    g0.y = lds_addr;                                               // [63:32] lds_addr
    g0.z = (unsigned)ga;                                           // global_addr lo
    g0.w = ((unsigned)(ga >> 32) & 0x01FFFFFFu) | (2u << 30);      // addr hi | type=2
    v8u g1;
    g1.s0 = (1u << 16)                  // data_size = 1 -> 2 bytes
          | (pad_en << 20)
          | (pad_interval << 22)
          | (pad_amount << 25);
    g1.s1 = (tensor_d0 & 0xFFFFu) << 16;                           // dim0[15:0] @48
    g1.s2 = (tensor_d0 >> 16) | ((tensor_d1 & 0xFFFFu) << 16);     // dim0 hi | dim1 lo
    g1.s3 = (tensor_d1 >> 16) | ((tile_d0 & 0xFFFFu) << 16);       // dim1 hi | tile0
    g1.s4 = (tile_d1 & 0xFFFFu);                                   // tile1 | tile2=0
    g1.s5 = stride_d0;                                             // dim0_stride lo
    g1.s6 = 0u;                                                    // stride hi | dim1_stride lo
    g1.s7 = 0u;
    v4u g2; g2.x = 1u; g2.y = 0u; g2.z = 0u; g2.w = 0u;            // tensor_dim2=1
    v4u g3; g3.x = 0u; g3.y = 0u; g3.z = 0u; g3.w = 0u;
    asm volatile("tensor_load_to_lds %0, %1, %2, %3"
                 :: "s"(g0), "s"(g1), "s"(g2), "s"(g3)
                 : "memory");
}
__device__ __forceinline__ void wait_tensor0() {
#if defined(__has_builtin)
#if __has_builtin(__builtin_amdgcn_s_wait_tensorcnt)
    __builtin_amdgcn_s_wait_tensorcnt(0);
    return;
#endif
#endif
    asm volatile("s_wait_tensorcnt 0x0" ::: "memory");
}

// ---------------------------------------------------------------------------
// Kernel 0: per-pixel Weierstrass features -> feats[N][4] (f32)
// ---------------------------------------------------------------------------
__global__ void feat_kernel(const float* __restrict__ alpha_p,
                            const float* __restrict__ beta_p,
                            const float* __restrict__ gamma_p,
                            const float* __restrict__ freq,
                            const float* __restrict__ scale_p,
                            float* __restrict__ feats) {
    int p = blockIdx.x * blockDim.x + threadIdx.x;   // 0..65535
    float omega2p = 0.12f  + softplus_f(alpha_p[0]);
    float beta    = 0.015f + softplus_f(beta_p[0]);
    float gp      = 0.15f  + softplus_f(gamma_p[0]);

    int j = p >> 8;     // column index (W)
    int i = p & 255;    // row index (H)
    float u = ((float)j + 0.5f) / (256.0f + 1e-8f);
    float v = ((float)i + 0.5f) / (256.0f + 1e-8f);

    float z_re = u * OMEGA1  + 0.025f * sinf(3.5f * PI_F * u);
    float z_im = v * omega2p + 0.025f * cosf(3.5f * PI_F * v);

    float z2 = z_re * z_re + z_im * z_im;
    float z_mod  = sqrtf(z2);
    float z_safe = fmaxf(z_mod, 1e-6f);
    float theta  = atan2f(z_im, z_re);
    float uu = z_re / OMEGA1;
    float vv = z_im / omega2p;

    const float kpow[4] = {1.0f, 0.28717459f, 0.13841807f, 0.08246924f};
    float au = fabsf(uu) * 0.8f;
    float av = fabsf(vv) * 0.8f;
    float corr = 0.0f, dcorr = 0.0f;
    #pragma unroll
    for (int k = 1; k <= 4; ++k) {
        float f   = (float)k * PI_F;
        float amp = gp * kpow[k - 1];
        float e_v = expf(-f * av);
        float e_u = expf(-f * au);
        float su  = sinf(f * uu), cu = cosf(f * uu);
        float sv  = sinf(f * vv), cv = cosf(f * vv);
        corr += amp * (cu * e_v + sv * e_u);
        if (k <= 3) dcorr += amp * (float)k * (-su * e_v + cv * e_u);
    }
    float ct = cosf(theta), st = sinf(theta);
    float main_  =  1.0f / (z_safe * z_safe + beta);
    float main_d = -2.0f / (z_safe * z_safe * z_safe + beta);
    float p_re  = main_  * ct + corr;
    float p_im  = main_  * st + 0.75f * corr;
    float pd_re = main_d * ct + dcorr;
    float pd_im = main_d * st + 0.6f  * dcorr;

    float s = scale_p[0];
    feats[p * 4 + 0] = tanhf(s * (p_re * freq[0]));
    feats[p * 4 + 1] = tanhf(s * (p_im * freq[1]) * 0.85f);
    feats[p * 4 + 2] = tanhf(s * pd_re * freq[2]);
    feats[p * 4 + 3] = tanhf(s * pd_im * freq[3]);
}

// ---------------------------------------------------------------------------
// Kernel 1: feats[N][4] @ W1[4][512] + b1 -> LN -> GELU -> h1[N][512] f16
// ---------------------------------------------------------------------------
__global__ __launch_bounds__(256) void mlp1_kernel(const float* __restrict__ feats,
                                                   const float* __restrict__ W1,
                                                   const float* __restrict__ b1,
                                                   const float* __restrict__ g1,
                                                   const float* __restrict__ be1,
                                                   _Float16* __restrict__ h1) {
    int wave = threadIdx.x >> 5;
    int lane = threadIdx.x & 31;
    int pix  = blockIdx.x * 8 + wave;

    float f0 = feats[pix * 4 + 0];
    float f1 = feats[pix * 4 + 1];
    float f2 = feats[pix * 4 + 2];
    float f3 = feats[pix * 4 + 3];

    float x[16];
    int jbase = lane * 16;
    float sum = 0.0f, sq = 0.0f;
    #pragma unroll
    for (int jj = 0; jj < 16; ++jj) {
        int j = jbase + jj;
        float a = b1[j] + f0 * W1[j] + f1 * W1[512 + j] + f2 * W1[1024 + j] + f3 * W1[1536 + j];
        x[jj] = a;
        sum += a;
        sq  += a * a;
    }
    #pragma unroll
    for (int m = 16; m >= 1; m >>= 1) {
        sum += __shfl_xor(sum, m, 32);
        sq  += __shfl_xor(sq,  m, 32);
    }
    float mu  = sum * (1.0f / 512.0f);
    float var = sq  * (1.0f / 512.0f) - mu * mu;
    float rs  = rsqrtf(var + 1e-5f);
    #pragma unroll
    for (int jj = 0; jj < 16; ++jj) {
        int j = jbase + jj;
        float t = (x[jj] - mu) * rs * g1[j] + be1[j];
        h1[pix * 512 + j] = (_Float16)gelu_exact(t);
    }
}

// ---------------------------------------------------------------------------
// Pack f32 row-major weight W[K][Nn] into WMMA-B fragment order (f16).
// ---------------------------------------------------------------------------
__global__ void pack_b_kernel(const float* __restrict__ W,
                              _Float16* __restrict__ Bp,
                              int K, int Nn) {
    int tid = blockIdx.x * blockDim.x + threadIdx.x;
    int ntiles = (K >> 5) * (Nn >> 4);
    if (tid >= ntiles * 32) return;
    int lane = tid & 31;
    int tile = tid >> 5;
    int ntile_cols = Nn >> 4;
    int nt = tile % ntile_cols;
    int kt = tile / ntile_cols;
    int n = nt * 16 + (lane & 15);
    int kbase = kt * 32 + (lane >> 4) * 16;
    v16h val;
    #pragma unroll
    for (int e = 0; e < 16; ++e)
        val[e] = (_Float16)W[(kbase + e) * Nn + n];
    ((v16h*)Bp)[tile * 32 + lane] = val;
}

// ---------------------------------------------------------------------------
// Kernel 2: h1[N][512] @ W2 -> +b2 -> LN(g2,be2) -> GELU -> h2[N][1024] f16
// A tile (16x512, contiguous 16KB) staged by TDM with the LDS pad engine
// producing the 520-half padded rows directly. B batched per K-step.
// ---------------------------------------------------------------------------
__global__ __launch_bounds__(256) void gemm2_kernel(const _Float16* __restrict__ h1,
                                                    const _Float16* __restrict__ Bp,
                                                    const float* __restrict__ b2,
                                                    const float* __restrict__ g2,
                                                    const float* __restrict__ be2,
                                                    _Float16* __restrict__ h2) {
    __shared__ _Float16 As[16 * 520];          // 512 + 8 pad (16B rows)
    __shared__ float redsum[16], redsq[16];

    int tid  = threadIdx.x;
    int wave = tid >> 5;
    int lane = tid & 31;
    int lg   = lane >> 4;
    int ln   = lane & 15;
    int row0 = blockIdx.x * 16;

    if (wave == 0) {
        // 16*512 = 8192 contiguous halfs; pad 4 DWORDs after every 256 DWORDs
        // (pad_interval=7 -> 256 DW = 512 halfs, pad_amount=3 -> 4 DW = 8 halfs)
        tdm_load_to_lds((unsigned)(uintptr_t)&As[0],
                        h1 + (size_t)row0 * 512,
                        /*tile*/8192, 1, /*tensor*/8192, 1, /*stride*/8192,
                        /*pad*/1, 7, 3);
        wait_tensor0();
    }
    if (tid < 16) { redsum[tid] = 0.0f; redsq[tid] = 0.0f; }
    __syncthreads();

    v8f acc[8];
    #pragma unroll
    for (int t = 0; t < 8; ++t)
        acc[t] = (v8f){0.f,0.f,0.f,0.f,0.f,0.f,0.f,0.f};

    const v16h* B = (const v16h*)Bp;
    for (int kt = 0; kt < 16; ++kt) {
        v16h a;
        int k0 = kt * 32 + lg * 16;
        v8h* ap = (v8h*)&a;
        ap[0] = *(const v8h*)(As + ln * 520 + k0);
        ap[1] = *(const v8h*)(As + ln * 520 + k0 + 8);
        v16h bf[8];
        #pragma unroll
        for (int t = 0; t < 8; ++t)
            bf[t] = B[((size_t)kt * 64 + wave * 8 + t) * 32 + lane];
        if (kt < 15)
            __builtin_prefetch(&B[((size_t)(kt + 1) * 64 + wave * 8) * 32 + lane], 0, 1);
        #pragma unroll
        for (int t = 0; t < 8; ++t)
            acc[t] = __builtin_amdgcn_wmma_f32_16x16x32_f16(
                false, a, false, bf[t], (short)0, acc[t], false, false);
    }

    float bn[8];
    #pragma unroll
    for (int t = 0; t < 8; ++t) bn[t] = b2[wave * 128 + t * 16 + ln];
    #pragma unroll
    for (int t = 0; t < 8; ++t) acc[t] = acc[t] + bn[t];

    #pragma unroll
    for (int r = 0; r < 8; ++r) {
        float ps = 0.0f, pq = 0.0f;
        #pragma unroll
        for (int t = 0; t < 8; ++t) { float x = acc[t][r]; ps += x; pq += x * x; }
        ps += __shfl_xor(ps, 8, 32);  pq += __shfl_xor(pq, 8, 32);
        ps += __shfl_xor(ps, 4, 32);  pq += __shfl_xor(pq, 4, 32);
        ps += __shfl_xor(ps, 2, 32);  pq += __shfl_xor(pq, 2, 32);
        ps += __shfl_xor(ps, 1, 32);  pq += __shfl_xor(pq, 1, 32);
        if (ln == 0) {
            int m = lg * 8 + r;
            atomicAdd(&redsum[m], ps);
            atomicAdd(&redsq[m],  pq);
        }
    }
    __syncthreads();

    float mu_r[8], rs_r[8];
    #pragma unroll
    for (int r = 0; r < 8; ++r) {
        int m = lg * 8 + r;
        float mu  = redsum[m] * (1.0f / 1024.0f);
        float var = redsq[m]  * (1.0f / 1024.0f) - mu * mu;
        mu_r[r] = mu;
        rs_r[r] = rsqrtf(var + 1e-5f);
    }
    #pragma unroll
    for (int t = 0; t < 8; ++t) {
        int n = wave * 128 + t * 16 + ln;
        float gg = g2[n], bb = be2[n];
        #pragma unroll
        for (int r = 0; r < 8; ++r) {
            int m = lg * 8 + r;
            float x = (acc[t][r] - mu_r[r]) * rs_r[r] * gg + bb;
            h2[(size_t)(row0 + m) * 1024 + n] = (_Float16)gelu_exact(x);
        }
    }
}

// ---------------------------------------------------------------------------
// Kernel 3: h2[N][1024] @ W3 + b3 + offset -> out[N][1024] f32
// 64x256 tile / workgroup, 2x4 wave grid, K double-buffered in LDS.
// A slab: per-lane ASYNCcnt copies. B slab (16KB contiguous): one TDM
// descriptor issued by wave 0 (TENSORcnt). Both overlap the WMMAs.
// ---------------------------------------------------------------------------
__global__ __launch_bounds__(256) void gemm3_kernel(const _Float16* __restrict__ h2,
                                                    const _Float16* __restrict__ Bp,
                                                    const float* __restrict__ b3,
                                                    const float* __restrict__ off,
                                                    float* __restrict__ out) {
    __shared__ _Float16 Asl[2][64 * 40];    // 10 KB  (64 rows x 32 k, pad->40)
    __shared__ _Float16 Bsl[2][16 * 512];   // 32 KB  (16 col tiles x 512 halfs)

    int tid  = threadIdx.x;
    int wave = tid >> 5;
    int lane = tid & 31;
    int lg   = lane >> 4;
    int ln   = lane & 15;
    int wm   = wave & 1;
    int wn   = wave >> 1;
    int row0 = blockIdx.x * 64;
    int nt0  = blockIdx.y * 16;

    int ar = tid >> 2;
    int ac = (tid & 3) * 8;
    const _Float16* aSrc0 = h2 + (size_t)(row0 + ar) * 1024 + ac;
    unsigned aDst0 = (unsigned)(uintptr_t)(&Asl[0][ar * 40 + ac]);
    unsigned aDst1 = (unsigned)(uintptr_t)(&Asl[1][ar * 40 + ac]);

    const _Float16* bSrcBase = Bp + (size_t)nt0 * 512;     // + kt*64*512 per step
    unsigned bBuf0 = (unsigned)(uintptr_t)(&Bsl[0][0]);
    unsigned bBuf1 = (unsigned)(uintptr_t)(&Bsl[1][0]);

    v8f acc[2][4];
    #pragma unroll
    for (int rt = 0; rt < 2; ++rt)
        #pragma unroll
        for (int ct = 0; ct < 4; ++ct)
            acc[rt][ct] = (v8f){0.f,0.f,0.f,0.f,0.f,0.f,0.f,0.f};

    // prime buffer 0 (kt = 0)
    async_ld_b128(aDst0, aSrc0);
    if (wave == 0)
        tdm_load_to_lds(bBuf0, bSrcBase, 8192, 1, 8192, 1, 8192, 0, 0, 0);
    wait_async0();
    if (wave == 0) wait_tensor0();
    __syncthreads();

    for (int kt = 0; kt < 32; ++kt) {
        int cur = kt & 1;
        if (kt + 1 < 32) {
            unsigned ad = (cur == 0) ? aDst1 : aDst0;
            async_ld_b128(ad, aSrc0 + (kt + 1) * 32);
            if (wave == 0) {
                unsigned bd = (cur == 0) ? bBuf1 : bBuf0;
                tdm_load_to_lds(bd, bSrcBase + (size_t)(kt + 1) * 64 * 512,
                                8192, 1, 8192, 1, 8192, 0, 0, 0);
            }
        }
        v16h a[2];
        #pragma unroll
        for (int rt = 0; rt < 2; ++rt) {
            int m = (wm * 2 + rt) * 16 + ln;
            const _Float16* arow = &Asl[cur][m * 40 + lg * 16];
            v8h* ap = (v8h*)&a[rt];
            ap[0] = *(const v8h*)(arow);
            ap[1] = *(const v8h*)(arow + 8);
        }
        #pragma unroll
        for (int ct = 0; ct < 4; ++ct) {
            v16h b = *(const v16h*)(&Bsl[cur][(wn * 4 + ct) * 512 + lane * 16]);
            #pragma unroll
            for (int rt = 0; rt < 2; ++rt)
                acc[rt][ct] = __builtin_amdgcn_wmma_f32_16x16x32_f16(
                    false, a[rt], false, b, (short)0, acc[rt][ct], false, false);
        }
        wait_async0();
        if (wave == 0) wait_tensor0();
        __syncthreads();
    }

    #pragma unroll
    for (int ct = 0; ct < 4; ++ct) {
        int n = (nt0 + wn * 4 + ct) * 16 + ln;
        float add = b3[n] + off[n];
        #pragma unroll
        for (int rt = 0; rt < 2; ++rt) {
            int mbase = row0 + (wm * 2 + rt) * 16 + lg * 8;
            #pragma unroll
            for (int r = 0; r < 8; ++r)
                out[(size_t)(mbase + r) * 1024 + n] = acc[rt][ct][r] + add;
        }
    }
}

// ---------------------------------------------------------------------------
extern "C" void kernel_launch(void* const* d_in, const int* in_sizes, int n_in,
                              void* d_out, int out_size, void* d_ws, size_t ws_size,
                              hipStream_t stream) {
    (void)in_sizes; (void)n_in; (void)out_size; (void)ws_size;
    // 0:H 1:W 2:alpha 3:beta 4:gamma 5:freq_mod 6:scale 7:offset
    // 8:W1 9:b1 10:g1 11:be1 12:W2 13:b2 14:g2 15:be2 16:W3 17:b3
    const float* alpha = (const float*)d_in[2];
    const float* beta  = (const float*)d_in[3];
    const float* gamma = (const float*)d_in[4];
    const float* freq  = (const float*)d_in[5];
    const float* scale = (const float*)d_in[6];
    const float* off   = (const float*)d_in[7];
    const float* W1    = (const float*)d_in[8];
    const float* b1    = (const float*)d_in[9];
    const float* g1    = (const float*)d_in[10];
    const float* be1   = (const float*)d_in[11];
    const float* W2    = (const float*)d_in[12];
    const float* b2    = (const float*)d_in[13];
    const float* g2    = (const float*)d_in[14];
    const float* be2   = (const float*)d_in[15];
    const float* W3    = (const float*)d_in[16];
    const float* b3    = (const float*)d_in[17];

    char* ws = (char*)d_ws;
    float*    feats = (float*)ws;                                 //   1 MB
    _Float16* h1    = (_Float16*)(ws + ((size_t)1   << 20));      //  64 MB
    _Float16* h2    = (_Float16*)(ws + ((size_t)65  << 20));      // 128 MB
    _Float16* Bp2   = (_Float16*)(ws + ((size_t)193 << 20));      //   1 MB
    _Float16* Bp3   = (_Float16*)(ws + ((size_t)194 << 20));      //   2 MB

    pack_b_kernel<<<128, 256, 0, stream>>>(W2, Bp2, 512, 1024);
    pack_b_kernel<<<256, 256, 0, stream>>>(W3, Bp3, 1024, 1024);

    feat_kernel<<<256, 256, 0, stream>>>(alpha, beta, gamma, freq, scale, feats);
    mlp1_kernel<<<8192, 256, 0, stream>>>(feats, W1, b1, g1, be1, h1);
    gemm2_kernel<<<4096, 256, 0, stream>>>(h1, Bp2, b2, g2, be2, h2);
    gemm3_kernel<<<dim3(1024, 4), 256, 0, stream>>>(h2, Bp3, b3, off, (float*)d_out);
}
